// GEDasMatchLoss_66391604462013
// MI455X (gfx1250) — compile-verified, matrix operand order checked by reading, not run on hardware
//
#include <hip/hip_runtime.h>
#include <hip/hip_bf16.h>
#include <math.h>

// ---------------- problem constants (match reference) ----------------
constexpr int BB  = 4;      // batch
constexpr int N   = 64;     // N1 == N2
constexpr int NE  = 256;    // NE1 == NE2
constexpr int LL  = N * N;  // 4096
constexpr float CONT_W = 0.25f;
constexpr float DIS_W  = 0.50f;
constexpr float ALPHA  = 0.2f;
constexpr float BETA   = 30.0f;
constexpr int MAX_ITER = 50;
constexpr int SK_ITER  = 20;

typedef __attribute__((ext_vector_type(16))) _Float16 v16h_t;
typedef __attribute__((ext_vector_type(8)))  float    v8f_t;

// ---------------------------------------------------------------------
// 1) Feature prep: normalize continuous part, fold weights into gen side,
//    and emit f16 operands PRE-SWIZZLED into the WMMA per-lane register
//    layout (16x32 f16 A/B striping), so the GEMM kernel does pure
//    b128 vector loads with zero packing VALU.
//    Packed layout: [rowblock][lane 0..31][half 0..15], 512 halves/block.
//      lane = (row%16) + 16*(k in [8,16) or [24,32))
//      half = (k%8) + 8*(k >= 16)
// ---------------------------------------------------------------------
__global__ void prep_features(const float* __restrict__ genX, const float* __restrict__ genE,
                              const float* __restrict__ obsX, const float* __restrict__ obsE,
                              _Float16* __restrict__ AgN, _Float16* __restrict__ BoN,
                              _Float16* __restrict__ AgE, _Float16* __restrict__ BoE) {
  int t = blockIdx.x * blockDim.x + threadIdx.x;
  const int nN = BB * N, nEg = BB * NE;
  const float* src; _Float16* dst; int nc; float wc, wd; int r;
  if (t < nN)                    { r = t;                src = genX + r * 16; dst = AgN; nc = 8; wc = CONT_W; wd = DIS_W; }
  else if (t < 2 * nN)           { r = t - nN;           src = obsX + r * 16; dst = BoN; nc = 8; wc = 1.f;    wd = 1.f;  }
  else if (t < 2 * nN + nEg)     { r = t - 2 * nN;       src = genE + r * 8;  dst = AgE; nc = 4; wc = CONT_W; wd = DIS_W; }
  else if (t < 2 * nN + 2 * nEg) { r = t - 2 * nN - nEg; src = obsE + r * 8;  dst = BoE; nc = 4; wc = 1.f;    wd = 1.f;  }
  else return;
  float nrm = 0.f;
  for (int k = 0; k < nc; ++k) nrm += src[k] * src[k];
  nrm = fmaxf(sqrtf(nrm), 1e-12f);
  float inv = wc / nrm;
  _Float16* blk = dst + (size_t)(r >> 4) * 512;   // 32 lanes * 16 halves
  int lr = r & 15;
  for (int k = 0; k < 32; ++k) {
    float v = 0.f;
    if (k < nc)          v = src[k] * inv;   // weighted, normalized continuous
    else if (k < 2 * nc) v = src[k] * wd;    // weighted one-hot discrete
    int lane = lr + (((k >> 3) & 1) << 4);
    int half = (k & 7) + (((k >> 4) & 1) << 3);
    blk[lane * 16 + half] = (_Float16)v;
  }
}

// ---------------------------------------------------------------------
// 2) WMMA affinity GEMM: D[b,m,n] = sum_k A[b,m,k]*B[b,n,k] + bias
//    4 waves / block, one 16x16 tile per wave, K padded to 32,
//    operands are pre-swizzled -> one aligned v16h load each.
// ---------------------------------------------------------------------
__global__ __launch_bounds__(128) void wmma_aff(const v16h_t* __restrict__ Ap,
                                                const v16h_t* __restrict__ Bp,
                                                float* __restrict__ D,
                                                int tilesM, int tilesN, float bias) {
  int tile = blockIdx.x * 4 + (threadIdx.x >> 5);
  int lane = threadIdx.x & 31;
  int tn = tile % tilesN; tile /= tilesN;
  int tm = tile % tilesM; int b = tile / tilesM;
  int rowsA = tilesM * 16, rowsB = tilesN * 16;
  v16h_t a  = Ap[(size_t)(b * tilesM + tm) * 32 + lane];   // 32B vector load
  v16h_t bv = Bp[(size_t)(b * tilesN + tn) * 32 + lane];
  v8f_t c = {};
  c = __builtin_amdgcn_wmma_f32_16x16x32_f16(false, a, false, bv, (short)0, c, false, false);
  int n = tn * 16 + (lane & 15);
  int mbase = tm * 16 + (lane >> 4) * 8;
  float* Dp = D + ((size_t)b * rowsA + mbase) * rowsB + n;
#pragma unroll
  for (int r = 0; r < 8; ++r)
    Dp[(size_t)r * rowsB] = c[r] + bias;
}

// ---------------------------------------------------------------------
// 3) Dedup masks: edge e survives iff no later edge has identical endpoints
//    (reproduces scatter-set "last write wins": winner = (max e1, max e2)).
// ---------------------------------------------------------------------
__global__ void keep_masks(const int* __restrict__ genA, const int* __restrict__ obsA,
                           int* __restrict__ keep1, int* __restrict__ keep2) {
  int t = blockIdx.x * blockDim.x + threadIdx.x;
  const int tot = BB * NE;
  const int* A; int* K; int idx;
  if (t < tot)          { A = genA; K = keep1; idx = t; }
  else if (t < 2 * tot) { A = obsA; K = keep2; idx = t - tot; }
  else return;
  int b = idx / NE, e = idx % NE;
  const int* Ab = A + b * 2 * NE;
  int a0 = Ab[e], a1 = Ab[NE + e];
  int kp = 1;
  for (int e2 = e + 1; e2 < NE; ++e2)
    if (Ab[e2] == a0 && Ab[NE + e2] == a1) { kp = 0; break; }
  K[idx] = kp;
}

// ---------------------------------------------------------------------
// 4) Row sums of K (diag = node_aff[i][j] at l=j*64+i; off-diag = kept edges)
// ---------------------------------------------------------------------
__global__ void rowsum_init(const float* __restrict__ node_aff, float* __restrict__ rs) {
  int t = blockIdx.x * blockDim.x + threadIdx.x;
  if (t >= BB * LL) return;
  int b = t >> 12, l = t & (LL - 1);
  rs[t] = node_aff[b * LL + (l & 63) * 64 + (l >> 6)];
}

__global__ void rowsum_scatter(const float* __restrict__ edge_aff,
                               const int* __restrict__ genA, const int* __restrict__ obsA,
                               const int* __restrict__ keep1, const int* __restrict__ keep2,
                               float* __restrict__ rs) {
  int t = blockIdx.x * blockDim.x + threadIdx.x;
  if (t >= BB * NE * NE) return;
  int b = t >> 16, p = t & 65535, e1 = p >> 8, e2 = p & 255;
  if (!keep1[b * NE + e1] || !keep2[b * NE + e2]) return;
  const int* g = genA + b * 2 * NE; const int* o = obsA + b * 2 * NE;
  int row = o[e2] * N + g[e1];
  int col = o[NE + e2] * N + g[NE + e1];
  if (row == col) return;                // diagonal is overwritten by node_aff
  atomicAdd(&rs[b * LL + row], edge_aff[(size_t)b * NE * NE + p]);
}

__global__ __launch_bounds__(1024) void dmax_kernel(const float* __restrict__ rs, float* __restrict__ dmax) {
  __shared__ float red[1024];
  int b = blockIdx.x, tid = threadIdx.x;
  float m = -3.4e38f;
  for (int l = tid; l < LL; l += 1024) m = fmaxf(m, rs[b * LL + l]);
  red[tid] = m; __syncthreads();
  for (int s = 512; s > 0; s >>= 1) { if (tid < s) red[tid] = fmaxf(red[tid], red[tid + s]); __syncthreads(); }
  if (tid == 0) dmax[b] = red[0];
}

// ---------------------------------------------------------------------
// 5) Fused RRWM: all 50 iterations per batch in one block, state in LDS.
//    Sparse K@v via ds_add_f32 atomics; log-Sinkhorn 20 iters on 64x64.
// ---------------------------------------------------------------------
__global__ __launch_bounds__(1024) void rrwm_kernel(const float* __restrict__ node_aff,
                                                    const float* __restrict__ edge_aff,
                                                    const int* __restrict__ genA, const int* __restrict__ obsA,
                                                    const int* __restrict__ keep1, const int* __restrict__ keep2,
                                                    const float* __restrict__ dmax,
                                                    float* __restrict__ soft) {
  __shared__ float u[LL], v[LL], slog[LL];
  __shared__ float red[1024];
  __shared__ int c1a[NE], c1b[NE], c2a[NE], c2b[NE], k1[NE], k2[NE];
  int b = blockIdx.x, tid = threadIdx.x;
  for (int e = tid; e < NE; e += 1024) {
    c1a[e] = genA[b * 2 * NE + e];       c1b[e] = genA[b * 2 * NE + NE + e];
    c2a[e] = obsA[b * 2 * NE + e];       c2b[e] = obsA[b * 2 * NE + NE + e];
    k1[e]  = keep1[b * NE + e];          k2[e]  = keep2[b * NE + e];
  }
  const float kn = 1.0f / (dmax[b] + 1e-8f);
  const float* eaff = edge_aff + (size_t)b * NE * NE;
  for (int l = tid; l < LL; l += 1024) v[l] = 1.0f / LL;
  __syncthreads();

  for (int it = 0; it < MAX_ITER; ++it) {
    // u = K @ v : diagonal part
    for (int l = tid; l < LL; l += 1024)
      u[l] = node_aff[b * LL + (l & 63) * 64 + (l >> 6)] * v[l];
    __syncthreads();
    // sparse edge part
    for (int t = tid; t < NE * NE; t += 1024) {
      int e1 = t >> 8, e2 = t & 255;
      if (k1[e1] & k2[e2]) {
        int row = c2a[e2] * N + c1a[e1];
        int col = c2b[e2] * N + c1b[e1];
        if (row != col) atomicAdd(&u[row], eaff[t] * v[col]);
      }
    }
    __syncthreads();
    // scale by 1/(dmax+eps); reduce sum|x| and max(x)
    float sa = 0.f, mx = -3.4e38f;
    for (int l = tid; l < LL; l += 1024) {
      float x = u[l] * kn; u[l] = x;
      sa += fabsf(x); mx = fmaxf(mx, x);
    }
    red[tid] = sa; __syncthreads();
    for (int s = 512; s > 0; s >>= 1) { if (tid < s) red[tid] += red[tid + s]; __syncthreads(); }
    float S = red[0]; __syncthreads();
    red[tid] = mx; __syncthreads();
    for (int s = 512; s > 0; s >>= 1) { if (tid < s) red[tid] = fmaxf(red[tid], red[tid + s]); __syncthreads(); }
    float M = red[0]; __syncthreads();
    float invS = 1.0f / S;
    float bM = BETA / M;   // BETA * v1 / max(v1) == BETA * x / max(x)
    for (int l = tid; l < LL; l += 1024) {
      float x = u[l];
      u[l] = x * invS;                               // v1 (l1-normalized walk)
      slog[(l & 63) * 64 + (l >> 6)] = x * bM;       // s[i][j], i=l%64, j=l/64
    }
    __syncthreads();
    // log-domain Sinkhorn, alternating row (axis=2) / col (axis=1)
    for (int si = 0; si < SK_ITER; ++si) {
      if (tid < 64) {
        if ((si & 1) == 0) {
          float m = -3.4e38f;
          for (int j = 0; j < 64; ++j) m = fmaxf(m, slog[tid * 64 + j]);
          float se = 0.f;
          for (int j = 0; j < 64; ++j) se += __expf(slog[tid * 64 + j] - m);
          float lse = m + __logf(se);
          for (int j = 0; j < 64; ++j) slog[tid * 64 + j] -= lse;
        } else {
          float m = -3.4e38f;
          for (int i = 0; i < 64; ++i) m = fmaxf(m, slog[i * 64 + tid]);
          float se = 0.f;
          for (int i = 0; i < 64; ++i) se += __expf(slog[i * 64 + tid] - m);
          float lse = m + __logf(se);
          for (int i = 0; i < 64; ++i) slog[i * 64 + tid] -= lse;
        }
      }
      __syncthreads();
    }
    // reweighted jump + l1 normalize
    float sa2 = 0.f;
    for (int l = tid; l < LL; l += 1024) {
      float sv = __expf(slog[(l & 63) * 64 + (l >> 6)]);
      float vn = ALPHA * sv + (1.0f - ALPHA) * u[l];
      v[l] = vn; sa2 += fabsf(vn);
    }
    red[tid] = sa2; __syncthreads();
    for (int s = 512; s > 0; s >>= 1) { if (tid < s) red[tid] += red[tid + s]; __syncthreads(); }
    float inv2 = 1.0f / red[0]; __syncthreads();
    for (int l = tid; l < LL; l += 1024) v[l] *= inv2;
    __syncthreads();
  }
  // soft[i][j] = v[j*64+i]
  for (int l = tid; l < LL; l += 1024)
    soft[b * LL + (l & 63) * 64 + (l >> 6)] = v[l];
}

// ---------------------------------------------------------------------
// 6) Jonker-Volgenant LAP per batch (maximize soft -> cost = -soft).
//    Column-parallel relaxation; thread 0 does scalar bookkeeping.
// ---------------------------------------------------------------------
__global__ __launch_bounds__(128) void hungarian_kernel(const float* __restrict__ soft, int* __restrict__ roc) {
  __shared__ float cost[N * N];
  __shared__ float du[N + 1], dv[N + 1], minv[N + 1];
  __shared__ int p[N + 1], way[N + 1], used[N + 1];
  __shared__ int j0s, brks; __shared__ float deltas;
  int b = blockIdx.x, tid = threadIdx.x;
  for (int t = tid; t < N * N; t += 128) cost[t] = -soft[b * N * N + t];
  if (tid <= N) { du[tid] = 0.f; dv[tid] = 0.f; p[tid] = 0; way[tid] = 0; }
  __syncthreads();
  for (int i = 1; i <= N; ++i) {
    if (tid == 0) { p[0] = i; j0s = 0; }
    if (tid <= N) { minv[tid] = 1e30f; used[tid] = 0; }
    __syncthreads();
    while (true) {
      if (tid == 0) used[j0s] = 1;
      __syncthreads();
      int j0 = j0s; int i0 = p[j0];
      if (tid >= 1 && tid <= N && !used[tid]) {
        float cur = cost[(i0 - 1) * N + (tid - 1)] - du[i0] - dv[tid];
        if (cur < minv[tid]) { minv[tid] = cur; way[tid] = j0; }
      }
      __syncthreads();
      if (tid == 0) {
        float best = 1e30f; int j1 = 1;
        for (int j = 1; j <= N; ++j)
          if (!used[j] && minv[j] < best) { best = minv[j]; j1 = j; }
        deltas = best; j0s = j1; brks = (p[j1] == 0);
      }
      __syncthreads();
      float delta = deltas;
      if (tid <= N) {
        if (used[tid]) { du[p[tid]] += delta; dv[tid] -= delta; }
        else if (tid >= 1) minv[tid] -= delta;
      }
      __syncthreads();
      if (brks) break;
    }
    if (tid == 0) {
      int j0 = j0s;
      while (j0 != 0) { int j1 = way[j0]; p[j0] = p[j1]; j0 = j1; }
    }
    __syncthreads();
  }
  if (tid < N) roc[b * N + tid] = p[tid + 1] - 1;   // row assigned to column tid
}

// ---------------------------------------------------------------------
// 7) Score: vx^T K vx in factorized form; output = -score.
// ---------------------------------------------------------------------
__global__ void score_init(float* __restrict__ score) {
  if (threadIdx.x < BB) score[threadIdx.x] = 0.f;
}

__global__ void score_node(const float* __restrict__ node_aff, const int* __restrict__ roc,
                           float* __restrict__ score) {
  int t = threadIdx.x + blockIdx.x * blockDim.x;
  if (t >= BB * N) return;
  int b = t >> 6, j = t & 63, i = roc[t];
  atomicAdd(&score[b], node_aff[b * LL + i * N + j]);
}

__global__ void score_edge(const float* __restrict__ edge_aff,
                           const int* __restrict__ genA, const int* __restrict__ obsA,
                           const int* __restrict__ keep1, const int* __restrict__ keep2,
                           const int* __restrict__ roc, float* __restrict__ score) {
  int t = blockIdx.x * blockDim.x + threadIdx.x;
  if (t >= BB * NE * NE) return;
  int b = t >> 16, p = t & 65535, e1 = p >> 8, e2 = p & 255;
  if (!keep1[b * NE + e1] || !keep2[b * NE + e2]) return;
  const int* g = genA + b * 2 * NE; const int* o = obsA + b * 2 * NE;
  int row = o[e2] * N + g[e1];
  int col = o[NE + e2] * N + g[NE + e1];
  if (row == col) return;
  if (roc[b * N + (row >> 6)] != (row & 63)) return;  // vx[row]
  if (roc[b * N + (col >> 6)] != (col & 63)) return;  // vx[col]
  atomicAdd(&score[b], edge_aff[(size_t)b * NE * NE + p]);
}

__global__ void finalize(const float* __restrict__ score, float* __restrict__ out) {
  if (threadIdx.x < BB) out[threadIdx.x] = -score[threadIdx.x];
}

// ---------------------------------------------------------------------
extern "C" void kernel_launch(void* const* d_in, const int* in_sizes, int n_in,
                              void* d_out, int out_size, void* d_ws, size_t ws_size,
                              hipStream_t stream) {
  (void)in_sizes; (void)n_in; (void)out_size; (void)ws_size;
  const float* genX = (const float*)d_in[0];
  const int*   genA = (const int*)  d_in[1];
  const float* genE = (const float*)d_in[2];
  const float* obsX = (const float*)d_in[3];
  const int*   obsA = (const int*)  d_in[4];
  const float* obsE = (const float*)d_in[5];

  char* ws = (char*)d_ws; size_t o = 0;
  _Float16* AgN = (_Float16*)(ws + o); o += (size_t)BB * N  * 32 * 2;   // 16 KB (packed: B*4 blocks)
  _Float16* BoN = (_Float16*)(ws + o); o += (size_t)BB * N  * 32 * 2;   // 16 KB
  _Float16* AgE = (_Float16*)(ws + o); o += (size_t)BB * NE * 32 * 2;   // 64 KB (packed: B*16 blocks)
  _Float16* BoE = (_Float16*)(ws + o); o += (size_t)BB * NE * 32 * 2;   // 64 KB
  float* node_aff = (float*)(ws + o);  o += (size_t)BB * LL * 4;        // 64 KB
  float* edge_aff = (float*)(ws + o);  o += (size_t)BB * NE * NE * 4;   // 1 MB
  int*   keep1    = (int*)  (ws + o);  o += (size_t)BB * NE * 4;
  int*   keep2    = (int*)  (ws + o);  o += (size_t)BB * NE * 4;
  float* rs       = (float*)(ws + o);  o += (size_t)BB * LL * 4;
  float* dmaxp    = (float*)(ws + o);  o += 256;
  float* soft     = (float*)(ws + o);  o += (size_t)BB * LL * 4;
  int*   roc      = (int*)  (ws + o);  o += (size_t)BB * N * 4;
  float* score    = (float*)(ws + o);  o += 256;

  prep_features<<<(2 * BB * N + 2 * BB * NE + 255) / 256, 256, 0, stream>>>(
      genX, genE, obsX, obsE, AgN, BoN, AgE, BoE);
  // node: 4x4 tiles/batch -> 64 tiles; edge: 16x16 tiles/batch -> 1024 tiles; 4 tiles/block
  wmma_aff<<<BB * 4 * 4 / 4, 128, 0, stream>>>((const v16h_t*)AgN, (const v16h_t*)BoN,
                                               node_aff, 4, 4, -(CONT_W + DIS_W));
  wmma_aff<<<BB * 16 * 16 / 4, 128, 0, stream>>>((const v16h_t*)AgE, (const v16h_t*)BoE,
                                                 edge_aff, 16, 16, -(CONT_W + DIS_W));
  keep_masks<<<(2 * BB * NE + 255) / 256, 256, 0, stream>>>(genA, obsA, keep1, keep2);
  rowsum_init<<<(BB * LL + 255) / 256, 256, 0, stream>>>(node_aff, rs);
  rowsum_scatter<<<(BB * NE * NE + 255) / 256, 256, 0, stream>>>(edge_aff, genA, obsA, keep1, keep2, rs);
  dmax_kernel<<<BB, 1024, 0, stream>>>(rs, dmaxp);
  rrwm_kernel<<<BB, 1024, 0, stream>>>(node_aff, edge_aff, genA, obsA, keep1, keep2, dmaxp, soft);
  hungarian_kernel<<<BB, 128, 0, stream>>>(soft, roc);
  score_init<<<1, 32, 0, stream>>>(score);
  score_node<<<1, 256, 0, stream>>>(node_aff, roc, score);
  score_edge<<<(BB * NE * NE + 255) / 256, 256, 0, stream>>>(edge_aff, genA, obsA, keep1, keep2, roc, score);
  finalize<<<1, 32, 0, stream>>>(score, (float*)d_out);
}
